// BasicBlock_54193897341016
// MI455X (gfx1250) — matrix operand order repaired and checked by reading, not compiled, not run on hardware
//
#include <hip/hip_runtime.h>
#include <math.h>

typedef _Float16 v8h  __attribute__((ext_vector_type(8)));
typedef _Float16 v16h __attribute__((ext_vector_type(16)));
typedef float    v8f  __attribute__((ext_vector_type(8)));

#define N_PTS   200000
#define C_IN    32
#define C_OUT   32
#define KOFF    27
#define BN_EPS  1e-5f
#define NTILES  (N_PTS / 16)                                   // 12500, exact
#define WAVES_PER_BLOCK 8                                      // 256 threads = 8 wave32
#define NBLK    ((NTILES + WAVES_PER_BLOCK - 1) / WAVES_PER_BLOCK)  // 1563

// ---------------- workspace layout (bytes, 256-aligned) ----------------
#define OFF_FEATH    0UL                      // N*32 f16            = 12,800,000
#define OFF_WB       12800000UL               // 27*2*32*16 f16      =     55,296
#define OFF_CONV     12855296UL               // N*32 f32            = 25,600,000
#define OFF_PART     38455296UL               // NBLK*64 f32         =    400,128
#define OFF_SB       38855424UL               // 64 f32

#define SB_HALVES   (KOFF * 2 * 32 * 16)      // 27648 halves = 55296 B
#define SB_UINT4    (SB_HALVES * 2 / 16)      // 3456 uint4

// 1) features f32 -> f16 (halves the random-gather traffic, enables f16 WMMA)
__global__ void feat_to_f16(const float* __restrict__ f, _Float16* __restrict__ h, int n) {
    int i = blockIdx.x * blockDim.x + threadIdx.x;
    if (i < n) h[i] = (_Float16)f[i];
}

// 2) weight [K][Cin][Cout] f32 -> WMMA B-tile layout, f16.
//    Tile t = k*2 + couthalf; lane L holds cout = half*16 + (L&15),
//    lanes 0-15: cin {0..7,16..23}, lanes 16-31: cin {8..15,24..31} (16 halves per lane).
__global__ void weight_prep(const float* __restrict__ w, _Float16* __restrict__ wB) {
    int k    = blockIdx.x >> 1;
    int half = blockIdx.x & 1;
    int lane = threadIdx.x;            // 0..31
    int cout  = half * 16 + (lane & 15);
    int kbase = (lane >> 4) * 16;
    _Float16* dst = wB + ((size_t)blockIdx.x * 32 + lane) * 16;
#pragma unroll
    for (int i = 0; i < 16; ++i) {
        int cin = kbase / 2 + i + (i >= 8 ? 8 : 0);
        dst[i] = (_Float16)w[((size_t)k * C_IN + cin) * C_OUT + cout];
    }
}

// 3) gather + GEMM via v_wmma_f32_16x16x32_f16, one wave per 16-row tile,
//    B tiles staged in LDS, BN partial sums fused from the accumulators.
__global__ __launch_bounds__(256) void conv_wmma(
        const _Float16* __restrict__ featH,
        const _Float16* __restrict__ wB,
        const int* __restrict__ nidx,
        float* __restrict__ convbuf,
        float* __restrict__ partials) {
    __shared__ __align__(32) _Float16 sB[SB_HALVES];   // 55296 B: all 54 B tiles
    __shared__ float red [WAVES_PER_BLOCK * 32];       // per-wave channel sums
    __shared__ float redq[WAVES_PER_BLOCK * 32];       // per-wave channel sumsq

    // cooperative block-wide copy of the swizzled weights into LDS
    {
        const uint4* src = (const uint4*)wB;
        uint4* dst = (uint4*)sB;
        for (int i = threadIdx.x; i < SB_UINT4; i += 256) dst[i] = src[i];
    }
    __syncthreads();

    int wave = threadIdx.x >> 5;
    int lane = threadIdx.x & 31;
    int tile = blockIdx.x * WAVES_PER_BLOCK + wave;
    bool active = tile < NTILES;       // wave-uniform: EXEC all-ones at each WMMA

    int m  = lane & 15;                // matrix row owned by this lane
    int hi = lane >> 4;                // 0: K-chunks {0..7,16..23}; 1: {8..15,24..31}

    v8f c0 = {};                       // cout 0..15
    v8f c1 = {};                       // cout 16..31

    if (active) {
        size_t row = (size_t)tile * 16 + m;
        const int* idxp = nidx + row * KOFF;
        for (int k = 0; k < KOFF; ++k) {
            int nb = idxp[k];
            // prefetch next tap's neighbor row (random gather -> hide L2 latency)
            if (k + 1 < KOFF) {
                int nb2 = idxp[k + 1];
                if (nb2 >= 0) __builtin_prefetch(featH + (size_t)nb2 * C_IN, 0, 1);
            }
            // A fragment: 16x32 f16, row m striped per ISA layout
            v16h a = {};
            if (nb >= 0) {
                const _Float16* fp = featH + (size_t)nb * C_IN + hi * 8;
                v8h p0 = *(const v8h*)(fp);        // K = hi?8..15 : 0..7
                v8h p1 = *(const v8h*)(fp + 16);   // K = hi?24..31: 16..23
#pragma unroll
                for (int i = 0; i < 8; ++i) { a[i] = p0[i]; a[8 + i] = p1[i]; }
            }
            const _Float16* wb = sB + ((size_t)(k * 2) * 32 + lane) * 16;
            v16h b0 = *(const v16h*)(wb);
            v16h b1 = *(const v16h*)(wb + 32 * 16);
            c0 = __builtin_amdgcn_wmma_f32_16x16x32_f16(false, a, false, b0, (short)0, c0, false, false);
            c1 = __builtin_amdgcn_wmma_f32_16x16x32_f16(false, a, false, b1, (short)0, c1, false, false);
        }
        // D layout: lane L -> col (L&15); VGPR v -> row v + (L>=16 ? 8 : 0)
        int rbase = tile * 16 + hi * 8;
#pragma unroll
        for (int v = 0; v < 8; ++v) {
            convbuf[(size_t)(rbase + v) * C_OUT + m]      = c0[v];
            convbuf[(size_t)(rbase + v) * C_OUT + 16 + m] = c1[v];
        }
    }

    // ---- fused BN partial statistics straight from the accumulators ----
    float s0 = 0.f, q0 = 0.f, s1 = 0.f, q1 = 0.f;
#pragma unroll
    for (int v = 0; v < 8; ++v) {
        s0 += c0[v]; q0 += c0[v] * c0[v];
        s1 += c1[v]; q1 += c1[v] * c1[v];
    }
    // lanes L and L+16 hold the same channel (different rows): merge
    s0 += __shfl_down(s0, 16); q0 += __shfl_down(q0, 16);
    s1 += __shfl_down(s1, 16); q1 += __shfl_down(q1, 16);
    if (lane < 16) {
        red [wave * 32 + lane]      = s0;   // channel = lane
        red [wave * 32 + 16 + lane] = s1;   // channel = 16 + lane
        redq[wave * 32 + lane]      = q0;
        redq[wave * 32 + 16 + lane] = q1;
    }
    __syncthreads();
    if (threadIdx.x < 32) {
        int c = threadIdx.x;
        float ts = 0.f, tq = 0.f;
#pragma unroll
        for (int w = 0; w < WAVES_PER_BLOCK; ++w) { ts += red[w * 32 + c]; tq += redq[w * 32 + c]; }
        partials[(size_t)blockIdx.x * 64 + c]      = ts;
        partials[(size_t)blockIdx.x * 64 + 32 + c] = tq;
    }
}

// 4) fold partials -> per-channel scale/bias
__global__ void bn_finalize(const float* __restrict__ partials,
                            const float* __restrict__ gamma,
                            const float* __restrict__ beta,
                            float* __restrict__ sb) {
    int c = threadIdx.x;
    if (c >= 32) return;
    float s = 0.f, q = 0.f;
    for (int b = 0; b < NBLK; ++b) {
        s += partials[(size_t)b * 64 + c];
        q += partials[(size_t)b * 64 + 32 + c];
    }
    float mean = s / (float)N_PTS;
    float var  = q / (float)N_PTS - mean * mean;   // biased, as torch BN
    float scl  = gamma[c] * rsqrtf(var + BN_EPS);
    sb[c]      = scl;
    sb[32 + c] = beta[c] - mean * scl;
}

// 5) fused BN apply + ReLU
__global__ void bn_apply(const float* __restrict__ convbuf,
                         const float* __restrict__ sb,
                         float* __restrict__ out) {
    int i = blockIdx.x * blockDim.x + threadIdx.x;
    if (i < N_PTS * C_OUT) {
        int c = i & 31;
        float x = fmaf(convbuf[i], sb[c], sb[32 + c]);
        out[i] = fmaxf(x, 0.f);
    }
}

extern "C" void kernel_launch(void* const* d_in, const int* in_sizes, int n_in,
                              void* d_out, int out_size, void* d_ws, size_t ws_size,
                              hipStream_t stream) {
    const float* features = (const float*)d_in[0];   // [N,32]
    const float* weight   = (const float*)d_in[1];   // [27,32,32]
    const float* gamma    = (const float*)d_in[2];   // [32]
    const float* beta     = (const float*)d_in[3];   // [32]
    const int*   nidx     = (const int*)d_in[4];     // [N,27]
    float* out = (float*)d_out;

    char* ws = (char*)d_ws;
    _Float16* featH   = (_Float16*)(ws + OFF_FEATH);
    _Float16* wB      = (_Float16*)(ws + OFF_WB);
    float*    convbuf = (float*)(ws + OFF_CONV);
    float*    partials= (float*)(ws + OFF_PART);
    float*    sb      = (float*)(ws + OFF_SB);

    int nfeat = N_PTS * C_IN;
    feat_to_f16<<<(nfeat + 255) / 256, 256, 0, stream>>>(features, featH, nfeat);
    weight_prep<<<KOFF * 2, 32, 0, stream>>>(weight, wB);

    conv_wmma<<<NBLK, 256, 0, stream>>>(featH, wB, nidx, convbuf, partials);

    bn_finalize<<<1, 32, 0, stream>>>(partials, gamma, beta, sb);

    int ntot = N_PTS * C_OUT;
    bn_apply<<<(ntot + 255) / 256, 256, 0, stream>>>(convbuf, sb, out);
}
